// AttentionXL_68788196212849
// MI455X (gfx1250) — matrix operand order repaired and checked by reading, not compiled, not run on hardware
//
#include <hip/hip_runtime.h>
#include <hip/hip_bf16.h>

typedef __attribute__((ext_vector_type(16))) __bf16        v16bf;
typedef __attribute__((ext_vector_type(8)))  float         v8f;
typedef __attribute__((ext_vector_type(16))) unsigned short v16u;
typedef __attribute__((ext_vector_type(8)))  unsigned int  v8u;
typedef __attribute__((ext_vector_type(4)))  int           v4i;

#define CS 1024
#define FS 2048
#define BSZ 4
#define DMODEL 1024
#define SROW 2052   // 16x2048 f32 score buffer, padded row stride (conflict-free)

#define AS1 __attribute__((address_space(1)))
#define AS3 __attribute__((address_space(3)))

#ifndef __has_builtin
#define __has_builtin(x) 0
#endif

#if __has_builtin(__builtin_amdgcn_global_load_async_to_lds_b128)
#define USE_ASYNC_LDS 1
#else
#define USE_ASYNC_LDS 0
#endif

static __device__ __forceinline__ void wait_async0() {
#if __has_builtin(__builtin_amdgcn_s_wait_asynccnt)
    __builtin_amdgcn_s_wait_asynccnt(0);
#else
    asm volatile("s_wait_asynccnt 0" ::: "memory");
#endif
}

static __device__ __forceinline__ unsigned short f2bf(float f) {
    unsigned int u = __float_as_uint(f);
    unsigned int r = (u + 0x7FFFu + ((u >> 16) & 1u)) >> 16;
    return (unsigned short)r;
}
static __device__ __forceinline__ float bf2f(unsigned short h) {
    return __uint_as_float(((unsigned int)h) << 16);
}

static __device__ __forceinline__ v16bf pack2(uint4 x, uint4 y) {
    v8u r;
    r[0] = x.x; r[1] = x.y; r[2] = x.z; r[3] = x.w;
    r[4] = y.x; r[5] = y.y; r[6] = y.z; r[7] = y.w;
    return __builtin_bit_cast(v16bf, r);
}

// A fragment (16x32 bf16): row fixed per lane; elements e0..7 -> K = 8g+e,
// e8..15 -> K = 16+8g+e.  rowp points at K=0 of this lane's row.
static __device__ __forceinline__ v16bf fragA(const unsigned short* rowp, int g) {
    uint4 x = *(const uint4*)(rowp + 8 * g);
    uint4 y = *(const uint4*)(rowp + 16 + 8 * g);
    return pack2(x, y);
}
// B fragment (32x16 bf16): col fixed per lane; elements e0..7 -> K = 8g+e, e8..15 -> K = 16+8g+e.
static __device__ __forceinline__ v16bf fragB(const unsigned short* colp, int g) {
    uint4 x = *(const uint4*)(colp + 8 * g);
    uint4 y = *(const uint4*)(colp + 16 + 8 * g);
    return pack2(x, y);
}

static __device__ __forceinline__ v8f wmma_bf16(v16bf a, v16bf b, v8f c) {
    return __builtin_amdgcn_wmma_f32_16x16x32_bf16(false, a, false, b, (short)0, c,
                                                   false, false);
}

// ---------------------------------------------------------------- convert f32->bf16
__global__ __launch_bounds__(256) void cvt_kernel(const float* __restrict__ in,
                                                  unsigned short* __restrict__ out,
                                                  int n) {
    int i = blockIdx.x * 256 + threadIdx.x;
    if (i < n) out[i] = f2bf(in[i]);
}

// ---------------------------------------------------------------- generic WMMA GEMM
// C(M,N) = A(MxK,row-major,bf16) * B(KxN,row-major,bf16), f32 accum.
// Block tile 128x64 (4 waves, each 32x64 -> 8 WMMA / K-step), double-buffered LDS,
// A tile staged with GLOBAL_LOAD_ASYNC_TO_LDS when available, B staged transposed.
// mode 0: out0[m*N+n] = bf16(acc + bias[n])                         (r projection)
// mode 1: outF[m*N+n] = acc + bias[n]                               (final output)
// mode 2: kv split: n<1024 -> key bf16 (row-major); n>=1024 -> V^T  (b,h,d,j) bf16
// mode 3: out0 = bf16(acc + bias[n] + u[n]); out1 = ... + v[n]      (q+u / q+v)
__global__ __launch_bounds__(128)
void gemm_bf16(const unsigned short* __restrict__ A,
               const unsigned short* __restrict__ B,
               const float* __restrict__ bias,
               const float* __restrict__ addU, const float* __restrict__ addV,
               unsigned short* __restrict__ out0, unsigned short* __restrict__ out1,
               float* __restrict__ outF,
               int M, int N, int K, int mode, int bsz, int fsz) {
    __shared__ __align__(16) unsigned short As[2][128][40];
    __shared__ __align__(16) unsigned short Bt[2][64][40];   // [n][k]

    const int tid = threadIdx.x;
    const int lane = tid & 31, wave = tid >> 5;
    const int g = lane >> 4, rA = lane & 15;
    const int m0 = blockIdx.y * 128;
    const int n0 = blockIdx.x * 64;
    const int NS = K / 32;

    v8f acc[2][4];
    for (int s = 0; s < 2; ++s)
        for (int t = 0; t < 4; ++t)
            acc[s][t] = (v8f){0.f, 0.f, 0.f, 0.f, 0.f, 0.f, 0.f, 0.f};

    uint4 breg[2];
#if !USE_ASYNC_LDS
    uint4 areg[4];
#endif

    auto loadB = [&](int k0) {
        for (int it = 0; it < 2; ++it) {
            int c = tid + 128 * it;
            int kk = c >> 3, nn = (c & 7) * 8;
            breg[it] = *(const uint4*)(B + (size_t)(k0 + kk) * N + n0 + nn);
        }
    };
    auto storeB = [&](int buf) {
        for (int it = 0; it < 2; ++it) {
            int c = tid + 128 * it;
            int kk = c >> 3, nn = (c & 7) * 8;
            unsigned short tmp[8];
            *(uint4*)tmp = breg[it];
            for (int e = 0; e < 8; ++e) Bt[buf][nn + e][kk] = tmp[e];
        }
    };
    auto issueA = [&](int k0, int buf) {
        for (int it = 0; it < 4; ++it) {
            int c = tid + 128 * it;
            int row = c >> 2, col = (c & 3) * 8;
#if USE_ASYNC_LDS
            __builtin_amdgcn_global_load_async_to_lds_b128(
                (AS1 v4i*)(unsigned long long)(A + (size_t)(m0 + row) * K + k0 + col),
                (AS3 v4i*)(&As[buf][row][col]), 0, 0);
#else
            areg[it] = *(const uint4*)(A + (size_t)(m0 + row) * K + k0 + col);
#endif
        }
    };
    auto commitA = [&](int buf) {
#if USE_ASYNC_LDS
        (void)buf;
        wait_async0();
#else
        for (int it = 0; it < 4; ++it) {
            int c = tid + 128 * it;
            int row = c >> 2, col = (c & 3) * 8;
            *(uint4*)(&As[buf][row][col]) = areg[it];
        }
#endif
    };

    // prologue: stage K-step 0 into buffer 0
    issueA(0, 0);
    loadB(0);
    storeB(0);
    commitA(0);
    __syncthreads();

    for (int s = 0; s < NS; ++s) {
        int cur = s & 1;
        if (s + 1 < NS) {            // prefetch next tile (overlaps WMMAs)
            issueA((s + 1) * 32, cur ^ 1);
            loadB((s + 1) * 32);
        }
        v16bf a0 = fragA(&As[cur][wave * 32 + rA][0], g);
        v16bf a1 = fragA(&As[cur][wave * 32 + 16 + rA][0], g);
        for (int t = 0; t < 4; ++t) {
            v16bf bf_ = fragB(&Bt[cur][t * 16 + rA][0], g);
            acc[0][t] = wmma_bf16(a0, bf_, acc[0][t]);
            acc[1][t] = wmma_bf16(a1, bf_, acc[1][t]);
        }
        if (s + 1 < NS) {
            storeB(cur ^ 1);
            commitA(cur ^ 1);
        }
        __syncthreads();
    }

    for (int su = 0; su < 2; ++su) {
        for (int t = 0; t < 4; ++t) {
            for (int r = 0; r < 8; ++r) {
                int a = r + 8 * g;
                int m = m0 + wave * 32 + su * 16 + a;
                int n = n0 + t * 16 + rA;
                float val = acc[su][t][r];
                if (mode == 0) {
                    out0[(size_t)m * N + n] = f2bf(val + bias[n]);
                } else if (mode == 1) {
                    outF[(size_t)m * N + n] = val + bias[n];
                } else if (mode == 2) {
                    val += bias[n];
                    int j = m / bsz, bb = m % bsz;
                    if (n < DMODEL) {
                        out0[(size_t)m * DMODEL + n] = f2bf(val);
                    } else {
                        int d2 = n - DMODEL;
                        out1[(((size_t)bb * 16 + (d2 >> 6)) * 64 + (d2 & 63)) * fsz +
                             j] = f2bf(val);
                    }
                } else {   // mode 3
                    float base = val + bias[n];
                    out0[(size_t)m * N + n] = f2bf(base + addU[n]);
                    out1[(size_t)m * N + n] = f2bf(base + addV[n]);
                }
            }
        }
    }
}

// ---------------------------------------------------------------- fused attention
// One workgroup per (16-row i-tile, b, h): content scores + rel-shifted position
// scores + softmax + PV, all through WMMA with a 16 x FS f32 score buffer in LDS.
__global__ __launch_bounds__(128)
void attn_kernel(const unsigned short* __restrict__ qu,
                 const unsigned short* __restrict__ qv,
                 const unsigned short* __restrict__ key,
                 const unsigned short* __restrict__ rbf,
                 const unsigned short* __restrict__ valT,
                 const unsigned char* __restrict__ mask,
                 unsigned short* __restrict__ av) {
    __shared__ __align__(16) float S[16 * SROW];
    __shared__ float red[16][8];

    const int tid = threadIdx.x;
    const int lane = tid & 31, wave = tid >> 5;
    const int g = lane >> 4, rA = lane & 15;
    const int i0 = blockIdx.x * 16;
    const int bh = blockIdx.y;
    const int b = bh >> 4, h = bh & 15;

    const unsigned short* quRow =
        qu + ((size_t)(i0 + rA) * BSZ + b) * DMODEL + h * 64;
    const unsigned short* qvRow =
        qv + ((size_t)(i0 + rA) * BSZ + b) * DMODEL + h * 64;
    v16bf aqu0 = fragA(quRow, g),  aqu1 = fragA(quRow + 32, g);
    v16bf aqv0 = fragA(qvRow, g),  aqv1 = fragA(qvRow + 32, g);

    // ---- pass 1: content scores (q+u) . k^T  ->  S
    for (int jt = wave; jt < FS / 16; jt += 4) {
        int j0 = jt * 16;
        const unsigned short* kcol =
            key + ((size_t)(j0 + rA) * BSZ + b) * DMODEL + h * 64;
        v8f acc = (v8f){0.f, 0.f, 0.f, 0.f, 0.f, 0.f, 0.f, 0.f};
        acc = wmma_bf16(aqu0, fragB(kcol, g), acc);
        acc = wmma_bf16(aqu1, fragB(kcol + 32, g), acc);
        for (int r = 0; r < 8; ++r) {
            int a = r + 8 * g;
            S[a * SROW + j0 + rA] = acc[r];
        }
    }
    __syncthreads();

    // ---- pass 2: position scores (q+v) . r^T, scattered by exact rel_shift flat math
    for (int jt = wave; jt < FS / 16; jt += 4) {
        int j0 = jt * 16;
        const unsigned short* rcol = rbf + (size_t)(j0 + rA) * DMODEL + h * 64;
        v8f acc = (v8f){0.f, 0.f, 0.f, 0.f, 0.f, 0.f, 0.f, 0.f};
        acc = wmma_bf16(aqv0, fragB(rcol, g), acc);
        acc = wmma_bf16(aqv1, fragB(rcol + 32, g), acc);
        for (int r = 0; r < 8; ++r) {
            int a2 = i0 + r + 8 * g;              // g row
            int t = j0 + rA;                      // g col (unshifted)
            int q_ = a2 * (FS + 1) + t + 1 - CS;  // flat index in padded buffer
            if (q_ >= 0) {
                int a = q_ / FS, c = q_ % FS;
                if (a >= i0 && a < i0 + 16) S[(a - i0) * SROW + c] += acc[r];
            }
        }
    }
    // extra g-row (i0+16): its t<shift part wraps into our last row
    {
        int a2x = i0 + 16;
        if (a2x < CS) {
            int shiftx = CS - 1 - a2x;
            const unsigned short* qrow =
                qv + ((size_t)a2x * BSZ + b) * DMODEL + h * 64;
            for (int t = tid; t < shiftx; t += 128) {
                const unsigned short* rrow = rbf + (size_t)t * DMODEL + h * 64;
                float s = 0.f;
                for (int d = 0; d < 64; ++d) s += bf2f(qrow[d]) * bf2f(rrow[d]);
                int q_ = a2x * (FS + 1) + t + 1 - CS;
                int a = q_ / FS, c = q_ % FS;
                if (a >= i0 && a < i0 + 16) S[(a - i0) * SROW + c] += s;
            }
        }
    }
    __syncthreads();

    // ---- softmax over j (scale, mask, stable): 8 threads per row x 256 cols
    {
        int row = tid >> 3, sub = tid & 7;
        const unsigned char* mrow = mask + (size_t)(i0 + row) * FS;
        float* srow = S + row * SROW;
        float mx = -__builtin_inff();
        for (int c = sub * 256; c < sub * 256 + 256; ++c) {
            float v = srow[c] * 0.125f;
            if (mrow[c]) v = -__builtin_inff();
            srow[c] = v;
            mx = fmaxf(mx, v);
        }
        red[row][sub] = mx;
        __syncthreads();
        float rmax = red[row][0];
        for (int k = 1; k < 8; ++k) rmax = fmaxf(rmax, red[row][k]);
        float sum = 0.f;
        for (int c = sub * 256; c < sub * 256 + 256; ++c) {
            float e = __expf(srow[c] - rmax);
            srow[c] = e;
            sum += e;
        }
        __syncthreads();
        red[row][sub] = sum;
        __syncthreads();
        float rsum = 0.f;
        for (int k = 0; k < 8; ++k) rsum += red[row][k];
        float inv = 1.0f / rsum;
        for (int c = sub * 256; c < sub * 256 + 256; ++c) srow[c] *= inv;
    }
    __syncthreads();

    // ---- PV: probs(16 x FS) @ V(FS x 64); each wave owns one 16-wide d tile
    {
        int nt = wave;
        const unsigned short* vrow =
            valT + (((size_t)b * 16 + h) * 64 + nt * 16 + rA) * FS;
        const float* srow = S + rA * SROW;
        v8f acc = (v8f){0.f, 0.f, 0.f, 0.f, 0.f, 0.f, 0.f, 0.f};
        for (int j0 = 0; j0 < FS; j0 += 32) {
            const float* p = srow + j0;
            float4 f0 = *(const float4*)(p + 8 * g);
            float4 f1 = *(const float4*)(p + 8 * g + 4);
            float4 f2 = *(const float4*)(p + 16 + 8 * g);
            float4 f3 = *(const float4*)(p + 16 + 8 * g + 4);
            v16u au;
            au[0] = f2bf(f0.x);  au[1] = f2bf(f0.y);  au[2] = f2bf(f0.z);  au[3] = f2bf(f0.w);
            au[4] = f2bf(f1.x);  au[5] = f2bf(f1.y);  au[6] = f2bf(f1.z);  au[7] = f2bf(f1.w);
            au[8] = f2bf(f2.x);  au[9] = f2bf(f2.y);  au[10] = f2bf(f2.z); au[11] = f2bf(f2.w);
            au[12] = f2bf(f3.x); au[13] = f2bf(f3.y); au[14] = f2bf(f3.z); au[15] = f2bf(f3.w);
            v16bf af = __builtin_bit_cast(v16bf, au);
            acc = wmma_bf16(af, fragB(vrow + j0, g), acc);
        }
        for (int r = 0; r < 8; ++r) {
            int a = r + 8 * g;
            int d = nt * 16 + rA;
            av[((size_t)(i0 + a) * BSZ + b) * DMODEL + h * 64 + d] = f2bf(acc[r]);
        }
    }
}

// ---------------------------------------------------------------- host launcher
extern "C" void kernel_launch(void* const* d_in, const int* in_sizes, int n_in,
                              void* d_out, int out_size, void* d_ws, size_t ws_size,
                              hipStream_t stream) {
    (void)in_sizes; (void)n_in; (void)out_size; (void)ws_size;
    const float* f_inputs = (const float*)d_in[0];
    const float* f_pos    = (const float*)d_in[1];
    const float* f_full   = (const float*)d_in[2];
    const float* f_u      = (const float*)d_in[3];
    const float* f_v      = (const float*)d_in[4];
    const unsigned char* f_mask = (const unsigned char*)d_in[5];
    const float* f_Wkv = (const float*)d_in[6];
    const float* f_bkv = (const float*)d_in[7];
    const float* f_Wq  = (const float*)d_in[8];
    const float* f_bq  = (const float*)d_in[9];
    const float* f_Wr  = (const float*)d_in[10];
    const float* f_br  = (const float*)d_in[11];
    const float* f_Wo  = (const float*)d_in[12];
    const float* f_bo  = (const float*)d_in[13];
    float* out = (float*)d_out;

    char* ws = (char*)d_ws;
    size_t off = 0;
    auto alloc = [&](size_t elems) -> unsigned short* {
        unsigned short* p = (unsigned short*)(ws + off);
        off += ((elems * 2 + 255) / 256) * 256;
        return p;
    };
    unsigned short* fi_bf  = alloc((size_t)FS * BSZ * DMODEL);     // full_input bf16
    unsigned short* wkv_bf = alloc((size_t)DMODEL * 2 * DMODEL);
    unsigned short* inp_bf = alloc((size_t)CS * BSZ * DMODEL);
    unsigned short* wq_bf  = alloc((size_t)DMODEL * DMODEL);
    unsigned short* pos_bf = alloc((size_t)FS * DMODEL);
    unsigned short* wr_bf  = alloc((size_t)DMODEL * DMODEL);
    unsigned short* wo_bf  = alloc((size_t)DMODEL * DMODEL);
    unsigned short* key_bf = alloc((size_t)FS * BSZ * DMODEL);     // (j,b,h*64+d)
    unsigned short* valT_bf= alloc((size_t)BSZ * 16 * 64 * FS);    // (b,h,d,j)
    unsigned short* qu_bf  = alloc((size_t)CS * BSZ * DMODEL);
    unsigned short* qv_bf  = alloc((size_t)CS * BSZ * DMODEL);
    unsigned short* r_bf   = alloc((size_t)FS * DMODEL);
    unsigned short* av_bf  = alloc((size_t)CS * BSZ * DMODEL);

    auto cvt = [&](const float* src, unsigned short* dst, size_t n) {
        cvt_kernel<<<(unsigned)((n + 255) / 256), 256, 0, stream>>>(src, dst, (int)n);
    };
    cvt(f_full,   fi_bf,  (size_t)FS * BSZ * DMODEL);
    cvt(f_Wkv,    wkv_bf, (size_t)DMODEL * 2 * DMODEL);
    cvt(f_inputs, inp_bf, (size_t)CS * BSZ * DMODEL);
    cvt(f_Wq,     wq_bf,  (size_t)DMODEL * DMODEL);
    cvt(f_pos,    pos_bf, (size_t)FS * DMODEL);
    cvt(f_Wr,     wr_bf,  (size_t)DMODEL * DMODEL);
    cvt(f_Wo,     wo_bf,  (size_t)DMODEL * DMODEL);

    dim3 blk(128);
    // kv = full_input @ Wkv + bkv  -> key (bf16) + V^T (bf16)
    gemm_bf16<<<dim3((2 * DMODEL) / 64, (FS * BSZ) / 128), blk, 0, stream>>>(
        fi_bf, wkv_bf, f_bkv, nullptr, nullptr, key_bf, valT_bf, nullptr,
        FS * BSZ, 2 * DMODEL, DMODEL, 2, BSZ, FS);
    // q = inputs @ Wq + bq  -> q+u, q+v (bf16)
    gemm_bf16<<<dim3(DMODEL / 64, (CS * BSZ) / 128), blk, 0, stream>>>(
        inp_bf, wq_bf, f_bq, f_u, f_v, qu_bf, qv_bf, nullptr,
        CS * BSZ, DMODEL, DMODEL, 3, BSZ, FS);
    // r = pos_embedding @ Wr + br (bf16)
    gemm_bf16<<<dim3(DMODEL / 64, FS / 128), blk, 0, stream>>>(
        pos_bf, wr_bf, f_br, nullptr, nullptr, r_bf, nullptr, nullptr,
        FS, DMODEL, DMODEL, 0, BSZ, FS);
    // fused attention
    attn_kernel<<<dim3(CS / 16, BSZ * 16), blk, 0, stream>>>(
        qu_bf, qv_bf, key_bf, r_bf, valT_bf, f_mask, av_bf);
    // out = attn_vec @ Wo + bo  (f32)
    gemm_bf16<<<dim3(DMODEL / 64, (CS * BSZ) / 128), blk, 0, stream>>>(
        av_bf, wo_bf, f_bo, nullptr, nullptr, nullptr, nullptr, out,
        CS * BSZ, DMODEL, DMODEL, 1, BSZ, FS);
}
